// Router_18090402251204
// MI455X (gfx1250) — matrix operand order repaired and checked by reading, not compiled, not run on hardware
//
#include <hip/hip_runtime.h>

typedef __attribute__((ext_vector_type(16))) __bf16 v16bf;
typedef __attribute__((ext_vector_type(8)))  __bf16 v8bf;
typedef __attribute__((ext_vector_type(8)))  float  v8f;

#define KC 256                    // K-chunk staged per iteration
#define LDS_STRIDE (KC + 8)       // bf16 elems; row stride 528 B -> 16B aligned, bank-shifted

// Issue all 32 global_load_b128 for one K-chunk into register arrays.
// Keeping them in one unrolled block lets the compiler put every load in
// flight before the first s_wait_loadcnt (max memory-level parallelism).
#define ISSUE_LOADS(K0)                                                        \
    _Pragma("unroll")                                                          \
    for (int it = 0; it < 16; ++it) {                                          \
        int i   = tid + it * 256;                                              \
        int row = i >> 6;                                                      \
        int c4  = i & 63;                                                      \
        xr[it] = *(const float4*)(x + (tok0 + row) * (long)D + (K0) + c4 * 4); \
        wr[it] = *(const float4*)(W + (long)row * D + (K0) + c4 * 4);          \
    }

// One workgroup: 64 tokens x 64 experts, 8 waves.
// Wave w: token tile = (w>>1)*16 .. +16, experts = (w&1)*32 .. +32 (two 16x16 tiles).
__global__ __launch_bounds__(256) void router_wmma_topk_kernel(
    const float* __restrict__ x,   // [T, D] fp32
    const float* __restrict__ W,   // [64, D] fp32
    const float* __restrict__ b,   // [64] fp32
    float* __restrict__ out_w,     // [T, 8] fp32
    int*   __restrict__ out_i,     // [T, 8] int32
    int D)
{
    __shared__ __bf16 xlds[64 * LDS_STRIDE];
    __shared__ __bf16 wlds[64 * LDS_STRIDE];
    __shared__ float  scores[64 * 65];        // padded to kill bank conflicts

    const int tid    = threadIdx.x;
    const int wave   = tid >> 5;              // 0..7 (wave32)
    const int lane   = tid & 31;
    const int l16    = lane & 15;
    const int half   = lane >> 4;             // 0: lanes 0-15, 1: lanes 16-31
    const int tokRow = (wave >> 1) * 16;      // token tile base within block
    const int eBase  = (wave & 1) * 32;       // expert base for this wave
    const long tok0  = (long)blockIdx.x * 64; // first token of this block

    v8f acc0 = {};
    v8f acc1 = {};

    float4 xr[16];                            // staged x chunk (64 VGPRs)
    float4 wr[16];                            // staged W chunk (64 VGPRs)

    ISSUE_LOADS(0)                            // prologue: loads for chunk 0 in flight

    for (int k0 = 0; k0 < D; k0 += KC) {
        __syncthreads();   // all waves done computing from LDS before we overwrite it

        // ---- drain staged registers: fp32 -> bf16, store to LDS ----
        #pragma unroll
        for (int it = 0; it < 16; ++it) {
            int i   = tid + it * 256;
            int row = i >> 6;                 // 0..63
            int c4  = i & 63;                 // float4 column
            __bf16* xd = &xlds[row * LDS_STRIDE + c4 * 4];
            xd[0] = (__bf16)xr[it].x; xd[1] = (__bf16)xr[it].y;
            xd[2] = (__bf16)xr[it].z; xd[3] = (__bf16)xr[it].w;
            __bf16* wd = &wlds[row * LDS_STRIDE + c4 * 4];
            wd[0] = (__bf16)wr[it].x; wd[1] = (__bf16)wr[it].y;
            wd[2] = (__bf16)wr[it].z; wd[3] = (__bf16)wr[it].w;
        }
        __syncthreads();

        // ---- prefetch next chunk's globals; they overlap the WMMA work below ----
        if (k0 + KC < D) {
            ISSUE_LOADS(k0 + KC)
        }

        // ---- WMMA across this K chunk (from LDS) ----
        const __bf16* arow  = &xlds[(tokRow + l16) * LDS_STRIDE];
        const __bf16* brow0 = &wlds[(eBase +      l16) * LDS_STRIDE];
        const __bf16* brow1 = &wlds[(eBase + 16 + l16) * LDS_STRIDE];
        #pragma unroll
        for (int kk = 0; kk < KC; kk += 32) {
            // A frag (16x32 bf16): lanes 0-15 row M=l16 hold K 0-15; lanes 16-31 hold K 16-31
            union { v16bf v; v8bf h[2]; } A, B0, B1;
            A.h[0]  = *(const v8bf*)(arow  + kk + 16 * half);
            A.h[1]  = *(const v8bf*)(arow  + kk + 16 * half + 8);
            // B frag (32x16 bf16): lanes 0-15 col N=l16 hold K 0-15; lanes 16-31 hold K 16-31
            B0.h[0] = *(const v8bf*)(brow0 + kk + 16 * half);
            B0.h[1] = *(const v8bf*)(brow0 + kk + 16 * half + 8);
            B1.h[0] = *(const v8bf*)(brow1 + kk + 16 * half);
            B1.h[1] = *(const v8bf*)(brow1 + kk + 16 * half + 8);
            acc0 = __builtin_amdgcn_wmma_f32_16x16x32_bf16(false, A.v, false, B0.v,
                                                           (short)0, acc0, false, false);
            acc1 = __builtin_amdgcn_wmma_f32_16x16x32_bf16(false, A.v, false, B1.v,
                                                           (short)0, acc1, false, false);
        }
    }
    __syncthreads();

    // ---- bias + sigmoid; scatter D tiles (VGPR v -> row v + 8*half, lane -> col) to LDS ----
    #pragma unroll
    for (int v = 0; v < 8; ++v) {
        int tl = tokRow + v + 8 * half;       // token row within block
        int e0 = eBase + l16;
        int e1 = eBase + 16 + l16;
        float l0 = acc0[v] + b[e0];
        float l1 = acc1[v] + b[e1];
        scores[tl * 65 + e0] = 1.0f / (1.0f + __expf(-l0));
        scores[tl * 65 + e1] = 1.0f / (1.0f + __expf(-l1));
    }
    __syncthreads();

    // ---- per-token top-8 over 64 experts (one thread per token) ----
    if (tid < 64) {
        float hv[8];
        int   hi[8];
        #pragma unroll
        for (int j = 0; j < 8; ++j) { hv[j] = -3.4e38f; hi[j] = 0; }
        for (int e = 0; e < 64; ++e) {
            float s = scores[tid * 65 + e];
            // fully-unrolled descending insertion; strict '>' keeps lowest index first on ties
            #pragma unroll
            for (int j = 7; j >= 1; --j) {
                if (s > hv[j]) {
                    if (s > hv[j - 1]) { hv[j] = hv[j - 1]; hi[j] = hi[j - 1]; }
                    else               { hv[j] = s;         hi[j] = e;         }
                }
            }
            if (s > hv[0]) { hv[0] = s; hi[0] = e; }
        }
        long token = tok0 + tid;
        #pragma unroll
        for (int j = 0; j < 8; ++j) {
            out_w[token * 8 + j] = hv[j];
            out_i[token * 8 + j] = hi[j];
        }
    }
}

extern "C" void kernel_launch(void* const* d_in, const int* in_sizes, int n_in,
                              void* d_out, int out_size, void* d_ws, size_t ws_size,
                              hipStream_t stream) {
    const float* x = (const float*)d_in[0];   // [B,S,D] fp32
    const float* W = (const float*)d_in[1];   // [E,D]   fp32
    const float* b = (const float*)d_in[2];   // [E]     fp32

    const int E = in_sizes[2];                // 64
    const int D = in_sizes[1] / E;            // 2048
    const int T = in_sizes[0] / D;            // 16384 tokens

    float* out_w = (float*)d_out;             // first output: top_w [T,8] fp32
    int*   out_i = (int*)d_out + (long)T * 8; // second output: top_idx [T,8] int32

    (void)n_in; (void)out_size; (void)d_ws; (void)ws_size;

    dim3 grid(T / 64);
    dim3 block(256);
    router_wmma_topk_kernel<<<grid, block, 0, stream>>>(x, W, b, out_w, out_i, D);
}
